// MambaBlock_23579370455199
// MI455X (gfx1250) — matrix operand order repaired and checked by reading, not compiled, not run on hardware
//
#include <hip/hip_runtime.h>

#define BSZ     4
#define SEQL    2048
#define DMODEL  1024
#define DINNER  2048
#define NH      32
#define HD      64
#define NSTATE  16
#define CONVD   2080
#define DPROJ   4160
#define NCHUNK  32
#define CHUNKL  64

typedef __attribute__((ext_vector_type(16))) __bf16 v16bf;
typedef __attribute__((ext_vector_type(8)))  __bf16 v8bf;
typedef __attribute__((ext_vector_type(8)))  float  v8f;

// ---------------------------------------------------------------------------
// WMMA helpers (wave32, ISA §7.12.2 layouts)
// ---------------------------------------------------------------------------
__device__ __forceinline__ v8f wmma_bf16(v16bf a, v16bf b, v8f c) {
  return __builtin_amdgcn_wmma_f32_16x16x32_bf16(false, a, false, b, (short)0, c,
                                                 false, false);
}

// A fragment, 16x32 bf16: lane row = M; low lanes K{0..7,16..23}, high lanes
// K{8..15,24..31}. base points at row start (16B aligned); hi = lane>>4.
__device__ __forceinline__ v16bf frag_a(const __bf16* base, int hi) {
  v8bf x0 = *(const v8bf*)(base + hi * 8);
  v8bf x1 = *(const v8bf*)(base + 16 + hi * 8);
  return __builtin_shufflevector(x0, x1, 0, 1, 2, 3, 4, 5, 6, 7,
                                 8, 9, 10, 11, 12, 13, 14, 15);
}

// B fragment, 32x16 bf16: lane col = N; low lanes K{0..7,16..23}, high lanes
// K{8..15,24..31} (same K order as frag_a, N-major row).
__device__ __forceinline__ v16bf frag_b(const __bf16* base, int hi) {
  v8bf x0 = *(const v8bf*)(base + hi * 8);
  v8bf x1 = *(const v8bf*)(base + 16 + hi * 8);
  return __builtin_shufflevector(x0, x1, 0, 1, 2, 3, 4, 5, 6, 7,
                                 8, 9, 10, 11, 12, 13, 14, 15);
}

// Async DMA: global (16B) -> LDS, tracked by ASYNCcnt (ISA §10.x / 08_async).
// vdst = LDS byte address VGPR, vaddr = 64-bit global address pair.
__device__ __forceinline__ void async_b128_to_lds(const void* gptr, void* lds) {
  asm volatile("global_load_async_to_lds_b128 %0, %1, off"
               :
               : "v"((unsigned)(unsigned long long)lds),
                 "v"((unsigned long long)gptr)
               : "memory");
}
__device__ __forceinline__ void wait_asynccnt0() {
  asm volatile("s_wait_asynccnt 0x0" ::: "memory");
}

// ---------------------------------------------------------------------------
// TN GEMM  C[M,N] = A[M,K] * B[N,K]^T  (bf16 WMMA, f32 accum)
// Block tile 256x64, BK=64, 256 threads = 8 waves (4x2), wave tile 64x32.
// Per LDS stage per wave: 16 v_wmma, 24 ds_load_b128, 1 barrier pair.
// ABF16 path stages the A tile with global_load_async_to_lds_b128 (no format
// conversion needed), synchronized via s_wait_asynccnt.
// ---------------------------------------------------------------------------
#define BMT 256
#define BNT 64
#define BKT 64
#define LDG 72   // 64 K + 8 pad bf16 -> 144B row stride (16B aligned)

template <bool ABF16>
__global__ __launch_bounds__(256) void gemm_tn(const void* __restrict__ Ain,
                                               const float* __restrict__ Bin,
                                               float* __restrict__ Cout,
                                               int M, int N, int K) {
  __shared__ __bf16 As[BMT * LDG];   // 36 KB
  __shared__ __bf16 Bs[BNT * LDG];   // 9 KB
  const int tid = threadIdx.x;
  const int lane = tid & 31, w = tid >> 5;
  const int wm = w >> 1, wn = w & 1;
  const int lrow = lane & 15, lhi = lane >> 4;
  const int mBase = blockIdx.y * BMT;
  const int nBase = blockIdx.x * BNT;
  v8f acc[4][2];
#pragma unroll
  for (int mi = 0; mi < 4; ++mi)
#pragma unroll
    for (int ni = 0; ni < 2; ++ni) acc[mi][ni] = (v8f){};

  for (int k0 = 0; k0 < K; k0 += BKT) {
    // ---- stage A tile (256 x 64) ----
    if constexpr (ABF16) {
      const __bf16* A = (const __bf16*)Ain;
#pragma unroll
      for (int i = 0; i < 8; ++i) {
        int idx = tid + i * 256;
        int r = idx >> 3, c8 = idx & 7;
        async_b128_to_lds(A + (size_t)(mBase + r) * K + k0 + c8 * 8,
                          &As[r * LDG + c8 * 8]);
      }
    } else {
      const float* A = (const float*)Ain;
#pragma unroll
      for (int i = 0; i < 16; ++i) {
        int idx = tid + i * 256;
        int r = idx >> 4, c4 = idx & 15;
        const float* gp = A + (size_t)(mBase + r) * K + k0 + c4 * 4;
        float4 v = *(const float4*)gp;
        __bf16* p = &As[r * LDG + c4 * 4];
        p[0] = (__bf16)v.x; p[1] = (__bf16)v.y;
        p[2] = (__bf16)v.z; p[3] = (__bf16)v.w;
        if (i == 0 && k0 + BKT < K)
          __builtin_prefetch(gp + BKT, 0, 1);   // global_prefetch_b8
      }
    }
    // ---- stage B tile (64 x 64) ----
#pragma unroll
    for (int i = 0; i < 4; ++i) {
      int idx = tid + i * 256;
      int r = idx >> 4, c4 = idx & 15;
      float4 v = *(const float4*)(Bin + (size_t)(nBase + r) * K + k0 + c4 * 4);
      __bf16* p = &Bs[r * LDG + c4 * 4];
      p[0] = (__bf16)v.x; p[1] = (__bf16)v.y;
      p[2] = (__bf16)v.z; p[3] = (__bf16)v.w;
    }
    if constexpr (ABF16) wait_asynccnt0();
    __syncthreads();

#pragma unroll
    for (int kk = 0; kk < 2; ++kk) {
      v16bf afr[4], bfr[2];
#pragma unroll
      for (int mi = 0; mi < 4; ++mi)
        afr[mi] = frag_a(&As[(wm * 64 + mi * 16 + lrow) * LDG + kk * 32], lhi);
#pragma unroll
      for (int ni = 0; ni < 2; ++ni)
        bfr[ni] = frag_b(&Bs[(wn * 32 + ni * 16 + lrow) * LDG + kk * 32], lhi);
#pragma unroll
      for (int mi = 0; mi < 4; ++mi)
#pragma unroll
        for (int ni = 0; ni < 2; ++ni)
          acc[mi][ni] = wmma_bf16(afr[mi], bfr[ni], acc[mi][ni]);
    }
    __syncthreads();
  }

#pragma unroll
  for (int mi = 0; mi < 4; ++mi)
#pragma unroll
    for (int ni = 0; ni < 2; ++ni) {
      int row0 = mBase + wm * 64 + mi * 16 + lhi * 8;
      int col = nBase + wn * 32 + ni * 16 + lrow;
#pragma unroll
      for (int i = 0; i < 8; ++i)
        Cout[(size_t)(row0 + i) * N + col] = acc[mi][ni][i];
    }
}

// ---------------------------------------------------------------------------
// depthwise causal conv (k=4) + bias + SiLU over xBC channels
// ---------------------------------------------------------------------------
__global__ __launch_bounds__(256) void conv_silu_kernel(
    const float* __restrict__ zx, const float* __restrict__ cw,
    const float* __restrict__ cb, float* __restrict__ xh,
    float* __restrict__ Bm, float* __restrict__ Cm) {
  size_t gid = (size_t)blockIdx.x * 256 + threadIdx.x;
  size_t total = (size_t)BSZ * SEQL * CONVD;
  if (gid >= total) return;
  int ch = (int)(gid % CONVD);
  size_t bt = gid / CONVD;
  int t = (int)(bt % SEQL);
  size_t b = bt / SEQL;
  float acc = cb[ch];
#pragma unroll
  for (int j = 0; j < 4; ++j) {
    int tt = t - 3 + j;
    if (tt >= 0)
      acc += zx[((size_t)b * SEQL + tt) * DPROJ + DINNER + ch] * cw[ch * 4 + j];
  }
  float v = acc / (1.f + __expf(-acc));  // silu
  size_t r = b * SEQL + t;
  if (ch < DINNER)            xh[r * DINNER + ch] = v;
  else if (ch < DINNER + 16)  Bm[r * NSTATE + (ch - DINNER)] = v;
  else                        Cm[r * NSTATE + (ch - DINNER - 16)] = v;
}

// ---------------------------------------------------------------------------
// dt = softplus(raw + bias), Adt = -exp(A_log)*dt
// ---------------------------------------------------------------------------
__global__ __launch_bounds__(256) void dt_kernel(const float* __restrict__ zx,
                                                 const float* __restrict__ dtb,
                                                 const float* __restrict__ Alog,
                                                 float* __restrict__ dt,
                                                 float* __restrict__ Adt) {
  size_t gid = (size_t)blockIdx.x * 256 + threadIdx.x;
  if (gid >= (size_t)BSZ * SEQL * NH) return;
  int h = (int)(gid & (NH - 1));
  size_t r = gid >> 5;
  float v = zx[r * DPROJ + (DINNER + CONVD) + h] + dtb[h];
  float sp = (v > 20.f) ? v : log1pf(__expf(v));
  dt[gid] = sp;
  Adt[gid] = -__expf(Alog[h]) * sp;
}

// ---------------------------------------------------------------------------
// per-(b,chunk,head) SSD: Y_diag + chunk states (all WMMA bf16)
// ---------------------------------------------------------------------------
__global__ __launch_bounds__(128) void ssd_chunk_kernel(
    const float* __restrict__ xh, const float* __restrict__ dt,
    const float* __restrict__ Adt, const float* __restrict__ Bm,
    const float* __restrict__ Cm, float* __restrict__ y,
    float* __restrict__ cstate, float* __restrict__ Asum_g,
    float* __restrict__ Acs_g) {
  const int bc = blockIdx.x;            // b*32 + c
  const int b = bc >> 5, c = bc & 31;
  const int h = blockIdx.y;
  const int tid = threadIdx.x;
  const int lane = tid & 31, w = tid >> 5;
  const int lrow = lane & 15, lhi = lane >> 4;

  __shared__ float sAdt[64], sAcs[64], sdt[64];
  __shared__ __bf16 sC[64 * 40];    // Cc rows l, K=n (cols 16..31 zero)
  __shared__ __bf16 sB[64 * 40];    // Bc rows s, K=n (cols 16..31 zero)
  __shared__ __bf16 sXT[64 * 72];   // xs^T rows p, cols l
  __shared__ __bf16 sM[64 * 72];    // M = G*L rows l, cols s
  __shared__ __bf16 sBd[16 * 72];   // (B*decay)^T rows n, cols l

  const size_t rowBase = (size_t)b * SEQL + (size_t)c * CHUNKL;
  if (tid < 64) {
    size_t r = rowBase + tid;
    sAdt[tid] = Adt[r * NH + h];
    sdt[tid] = dt[r * NH + h];
  }
  __syncthreads();
  if (tid == 0) {
    float s = 0.f;
    for (int l = 0; l < 64; ++l) { s += sAdt[l]; sAcs[l] = s; }
  }
  __syncthreads();
  const float Asum = sAcs[63];

  for (int i = tid; i < 64 * 16; i += 128) {
    int l = i >> 4, n = i & 15;
    size_t r = rowBase + l;
    float cv = Cm[r * NSTATE + n];
    float bv = Bm[r * NSTATE + n];
    sC[l * 40 + n] = (__bf16)cv;  sC[l * 40 + 16 + n] = (__bf16)0.f;
    sB[l * 40 + n] = (__bf16)bv;  sB[l * 40 + 16 + n] = (__bf16)0.f;
    sBd[n * 72 + l] = (__bf16)(bv * __expf(Asum - sAcs[l]));
  }
  for (int i = tid; i < 64 * 64; i += 128) {
    int l = i >> 6, p = i & 63;
    size_t r = rowBase + l;
    sXT[p * 72 + l] = (__bf16)(xh[r * DINNER + h * HD + p] * sdt[l]);
  }
  __syncthreads();

  // G = Cc * Bc^T (K=16 zero-padded to 32); M = G .* exp(seg) masked.
  // Branchless: exp always computed, mask applied with cndmask.
  {
    int lt = w;
    v16bf a = frag_a(&sC[(lt * 16 + lrow) * 40], lhi);
    for (int st = 0; st < 4; ++st) {
      v16bf bf = frag_b(&sB[(st * 16 + lrow) * 40], lhi);
      v8f g = {};
      g = wmma_bf16(a, bf, g);
#pragma unroll
      for (int i = 0; i < 8; ++i) {
        int l = lt * 16 + i + lhi * 8;
        int s = st * 16 + lrow;
        float e = __expf(sAcs[l] - sAcs[s]);
        e = (l >= s) ? e : 0.f;
        sM[l * 72 + s] = (__bf16)(g[i] * e);
      }
    }
  }
  __syncthreads();

  // Y_diag = M * xs   (64x64x64)
  {
    int lt = w;
    const __bf16* aRow = &sM[(lt * 16 + lrow) * 72];
    for (int pt = 0; pt < 4; ++pt) {
      v8f acc = {};
#pragma unroll
      for (int kk = 0; kk < 2; ++kk) {
        v16bf a = frag_a(aRow + kk * 32, lhi);
        v16bf bf = frag_b(&sXT[(pt * 16 + lrow) * 72 + kk * 32], lhi);
        acc = wmma_bf16(a, bf, acc);
      }
#pragma unroll
      for (int i = 0; i < 8; ++i) {
        int l = lt * 16 + i + lhi * 8;
        int p = pt * 16 + lrow;
        y[(rowBase + l) * DINNER + h * HD + p] = acc[i];
      }
    }
  }

  // chunk_state = xs^T * (B*decay)   (64x16x64)
  {
    int pt = w;
    v8f acc = {};
#pragma unroll
    for (int kk = 0; kk < 2; ++kk) {
      v16bf a = frag_a(&sXT[(pt * 16 + lrow) * 72 + kk * 32], lhi);
      v16bf bf = frag_b(&sBd[lrow * 72 + kk * 32], lhi);
      acc = wmma_bf16(a, bf, acc);
    }
    size_t base = ((size_t)bc * NH + h) * (HD * NSTATE);
#pragma unroll
    for (int i = 0; i < 8; ++i) {
      int p = pt * 16 + i + lhi * 8;
      cstate[base + p * NSTATE + lrow] = acc[i];
    }
  }

  if (tid == 0) Asum_g[bc * NH + h] = Asum;
  if (tid < 64) Acs_g[((size_t)bc * NH + h) * 64 + tid] = sAcs[tid];
}

// ---------------------------------------------------------------------------
// sequential inter-chunk state scan per (b,h)
// ---------------------------------------------------------------------------
__global__ __launch_bounds__(256) void ssd_scan_kernel(
    const float* __restrict__ cstate, const float* __restrict__ Asum_g,
    float* __restrict__ st_g) {
  int bh = blockIdx.x;
  int b = bh >> 5, h = bh & 31;
  int tid = threadIdx.x;
  float run[4] = {0.f, 0.f, 0.f, 0.f};
  for (int c = 0; c < NCHUNK; ++c) {
    int bc = b * NCHUNK + c;
    size_t base = ((size_t)bc * NH + h) * (HD * NSTATE);
    float dec = __expf(Asum_g[bc * NH + h]);
#pragma unroll
    for (int j = 0; j < 4; ++j) {
      size_t e = (size_t)tid * 4 + j;
      st_g[base + e] = run[j];
      run[j] = run[j] * dec + cstate[base + e];
    }
  }
}

// ---------------------------------------------------------------------------
// Y_off = exp(Acs) .* (Cc * st^T)  + D skip, accumulate into y
// ---------------------------------------------------------------------------
__global__ __launch_bounds__(128) void ssd_off_kernel(
    const float* __restrict__ Cm, const float* __restrict__ st_g,
    const float* __restrict__ Acs_g, const float* __restrict__ xh,
    const float* __restrict__ Dv, float* __restrict__ y) {
  const int bc = blockIdx.x;
  const int b = bc >> 5, c = bc & 31;
  const int h = blockIdx.y;
  const int tid = threadIdx.x;
  const int lane = tid & 31, w = tid >> 5;
  const int lrow = lane & 15, lhi = lane >> 4;

  __shared__ __bf16 sC[64 * 40];   // Cc rows l (cols 16..31 zero)
  __shared__ __bf16 sS[64 * 40];   // st rows p, cols n (cols 16..31 zero)
  __shared__ float sAcs[64];

  const size_t rowBase = (size_t)b * SEQL + (size_t)c * CHUNKL;
  size_t sbase = ((size_t)bc * NH + h) * (HD * NSTATE);
  for (int i = tid; i < 64 * 16; i += 128) {
    int l = i >> 4, n = i & 15;
    sC[l * 40 + n] = (__bf16)Cm[(rowBase + l) * NSTATE + n];
    sC[l * 40 + 16 + n] = (__bf16)0.f;
    sS[l * 40 + n] = (__bf16)st_g[sbase + l * NSTATE + n];  // here l == p
    sS[l * 40 + 16 + n] = (__bf16)0.f;
  }
  if (tid < 64) sAcs[tid] = Acs_g[((size_t)bc * NH + h) * 64 + tid];
  __syncthreads();

  int lt = w;
  v16bf a = frag_a(&sC[(lt * 16 + lrow) * 40], lhi);
  float Dh = Dv[h];
  for (int pt = 0; pt < 4; ++pt) {
    v16bf bf = frag_b(&sS[(pt * 16 + lrow) * 40], lhi);
    v8f acc = {};
    acc = wmma_bf16(a, bf, acc);
#pragma unroll
    for (int i = 0; i < 8; ++i) {
      int l = lt * 16 + i + lhi * 8;
      int p = pt * 16 + lrow;
      size_t idx = (rowBase + l) * DINNER + h * HD + p;
      y[idx] = y[idx] + acc[i] * __expf(sAcs[l]) + xh[idx] * Dh;
    }
  }
}

// ---------------------------------------------------------------------------
// gating (y * silu(z)) + RMSNorm, emit bf16 activations
// ---------------------------------------------------------------------------
__global__ __launch_bounds__(256) void gate_norm_kernel(
    const float* __restrict__ y, const float* __restrict__ zx,
    const float* __restrict__ nw, __bf16* __restrict__ yn) {
  size_t r = blockIdx.x;
  int tid = threadIdx.x;
  __shared__ float red[256];
  float g[8];
  float ss = 0.f;
#pragma unroll
  for (int j = 0; j < 8; ++j) {
    int e = j * 256 + tid;
    float z = zx[r * DPROJ + e];
    float v = y[r * DINNER + e] * (z / (1.f + __expf(-z)));
    g[j] = v;
    ss += v * v;
  }
  red[tid] = ss;
  __syncthreads();
  for (int s = 128; s > 0; s >>= 1) {
    if (tid < s) red[tid] += red[tid + s];
    __syncthreads();
  }
  float scale = rsqrtf(red[0] / (float)DINNER + 1e-5f);
#pragma unroll
  for (int j = 0; j < 8; ++j) {
    int e = j * 256 + tid;
    yn[r * DINNER + e] = (__bf16)(g[j] * scale * nw[e]);
  }
}

// ---------------------------------------------------------------------------
// Launch
// ---------------------------------------------------------------------------
extern "C" void kernel_launch(void* const* d_in, const int* in_sizes, int n_in,
                              void* d_out, int out_size, void* d_ws,
                              size_t ws_size, hipStream_t stream) {
  (void)in_sizes; (void)n_in; (void)out_size; (void)ws_size;
  const float* x      = (const float*)d_in[0];
  const float* W_in   = (const float*)d_in[1];
  const float* conv_w = (const float*)d_in[2];
  const float* conv_b = (const float*)d_in[3];
  const float* dt_b   = (const float*)d_in[4];
  const float* A_log  = (const float*)d_in[5];
  const float* Dv     = (const float*)d_in[6];
  const float* norm_w = (const float*)d_in[7];
  const float* W_out  = (const float*)d_in[8];
  float* out = (float*)d_out;

  float* ws = (float*)d_ws;
  size_t o = 0;
  float* zx   = ws + o; o += (size_t)BSZ * SEQL * DPROJ;
  float* xh   = ws + o; o += (size_t)BSZ * SEQL * DINNER;
  float* Bm   = ws + o; o += (size_t)BSZ * SEQL * NSTATE;
  float* Cm   = ws + o; o += (size_t)BSZ * SEQL * NSTATE;
  float* dtb  = ws + o; o += (size_t)BSZ * SEQL * NH;
  float* Adt  = ws + o; o += (size_t)BSZ * SEQL * NH;
  float* yb   = ws + o; o += (size_t)BSZ * SEQL * DINNER;
  float* cs   = ws + o; o += (size_t)BSZ * NCHUNK * NH * HD * NSTATE;
  float* st   = ws + o; o += (size_t)BSZ * NCHUNK * NH * HD * NSTATE;
  float* Asum = ws + o; o += (size_t)BSZ * NCHUNK * NH;
  float* Acs  = ws + o; o += (size_t)BSZ * NCHUNK * NH * CHUNKL;
  __bf16* yn  = (__bf16*)(ws + o);

  const int M = BSZ * SEQL;  // 8192

  gemm_tn<false><<<dim3(DPROJ / BNT, M / BMT), 256, 0, stream>>>(
      x, W_in, zx, M, DPROJ, DMODEL);
  conv_silu_kernel<<<((size_t)BSZ * SEQL * CONVD) / 256, 256, 0, stream>>>(
      zx, conv_w, conv_b, xh, Bm, Cm);
  dt_kernel<<<((size_t)BSZ * SEQL * NH) / 256, 256, 0, stream>>>(
      zx, dt_b, A_log, dtb, Adt);
  ssd_chunk_kernel<<<dim3(BSZ * NCHUNK, NH), 128, 0, stream>>>(
      xh, dtb, Adt, Bm, Cm, yb, cs, Asum, Acs);
  ssd_scan_kernel<<<BSZ * NH, 256, 0, stream>>>(cs, Asum, st);
  ssd_off_kernel<<<dim3(BSZ * NCHUNK, NH), 128, 0, stream>>>(
      Cm, st, Acs, xh, Dv, yb);
  gate_norm_kernel<<<M, 256, 0, stream>>>(yb, zx, norm_w, yn);
  gemm_tn<true><<<dim3(DMODEL / BNT, M / BMT), 256, 0, stream>>>(
      yn, W_out, out, M, DMODEL, DINNER);
}
